// PooledBert_23038204576438
// MI455X (gfx1250) — matrix operand order repaired and checked by reading, not compiled, not run on hardware
//
#include <hip/hip_runtime.h>
#include <hip/hip_bf16.h>
#include <math.h>

// ---------------- model constants ----------------
#define BATCH   4
#define SEQ     512
#define DMODEL  768
#define NHEADS  12
#define VOCAB   32000
#define ROWS    (BATCH * SEQ)          // 2048
#define FFN     (4 * DMODEL)           // 3072

// ---------------- WMMA GEMM tiling ----------------
#define BM 128
#define BN 128
#define BK 32
#define BKP 36                          // A row stride in LDS (u16): 32 + 2-dword TDM pad
#define BNP (BN + 2)                    // Bsp row stride in LDS (u32): 128 + 2-dword TDM pad

typedef __attribute__((ext_vector_type(16))) __bf16       bf16x16;
typedef __attribute__((ext_vector_type(8)))  unsigned int u32x8;
typedef __attribute__((ext_vector_type(8)))  float        f32x8;
// TDM descriptor groups (probe-verified builtin signatures)
typedef __attribute__((ext_vector_type(4)))  unsigned int tdm_g0_t;
typedef __attribute__((ext_vector_type(8)))  int          tdm_g1_t;
typedef __attribute__((ext_vector_type(4)))  int          tdm_g2_t;

// ---------------- helpers ----------------
__device__ __forceinline__ float blockReduceSum(float v, float* red) {
    int t = threadIdx.x;
    red[t] = v; __syncthreads();
    for (int s = 128; s > 0; s >>= 1) {
        if (t < s) red[t] += red[t + s];
        __syncthreads();
    }
    float r = red[0]; __syncthreads();
    return r;
}

__device__ __forceinline__ float blockReduceMax(float v, float* red) {
    int t = threadIdx.x;
    red[t] = v; __syncthreads();
    for (int s = 128; s > 0; s >>= 1) {
        if (t < s) red[t] = fmaxf(red[t], red[t + s]);
        __syncthreads();
    }
    float r = red[0]; __syncthreads();
    return r;
}

__device__ __forceinline__ unsigned short f32_to_bf16_rne(float f) {
    unsigned u = __float_as_uint(f);
    return (unsigned short)((u + 0x7FFFu + ((u >> 16) & 1u)) >> 16);
}

// LDS aperture places the LDS byte offset in addr[31:0] of the generic pointer.
__device__ __forceinline__ unsigned lds_addr_u32(const void* p) {
    return (unsigned)(unsigned long long)p;
}

// ---------------- TDM 2D tile load: global -> LDS, OOB zero-fill + row pad ----------------
// dsz_code: 0=1B,1=2B,2=4B ; pad_int_code: pad after 2^(code+1) dwords ; pad amount fixed 2 dwords
__device__ __forceinline__ void tdm_load_2d(unsigned lds_addr, const void* gptr,
                                            int rem0, int rem1, int tile0, int tile1,
                                            int stride0, int dsz_code, int pad_int_code) {
    unsigned long long ga = (unsigned long long)gptr;
    tdm_g0_t g0;
    g0[0] = 1u;                                   // count=1 valid descriptor
    g0[1] = lds_addr;                             // LDS byte address
    g0[2] = (unsigned)ga;                         // global_addr[31:0]
    g0[3] = ((unsigned)(ga >> 32) & 0x01FFFFFFu)  // global_addr[56:32]
            | (2u << 30);                         // type = 2 ("image")
    tdm_g1_t g1;
    g1[0] = (int)(((unsigned)dsz_code << 16)      // data_size
                  | (1u << 20)                    // pad_enable
                  | ((unsigned)pad_int_code << 22)// pad_interval
                  | (1u << 25));                  // pad_amount code 1 -> 2 dwords
    g1[1] = (int)(((unsigned)rem0 & 0xFFFFu) << 16);                       // tensor_dim0[15:0]
    g1[2] = (int)(((unsigned)rem0 >> 16) | (((unsigned)rem1 & 0xFFFFu) << 16)); // dim0 hi | dim1 lo
    g1[3] = (int)(((unsigned)rem1 >> 16) | ((unsigned)tile0 << 16));       // dim1 hi | tile_dim0
    g1[4] = (int)((unsigned)tile1);                                        // tile_dim1 (tile_dim2=0)
    g1[5] = (int)(unsigned)stride0;                                        // tensor_dim0_stride lo32
    g1[6] = 0;
    g1[7] = 0;
    tdm_g2_t gz = {0, 0, 0, 0};
#if __clang_major__ >= 23
    tdm_g1_t z8 = {0, 0, 0, 0, 0, 0, 0, 0};
    __builtin_amdgcn_tensor_load_to_lds(g0, g1, gz, gz, z8, 0);
#else
    __builtin_amdgcn_tensor_load_to_lds(g0, g1, gz, gz, 0);
#endif
}

// ---------------- elementwise: f32 -> bf16(u16), plain layout ----------------
__global__ void pb_cvt_bf16(const float* __restrict__ in,
                            unsigned short* __restrict__ out, size_t n) {
    size_t i = (size_t)blockIdx.x * 256 + threadIdx.x;
    if (i < n) out[i] = f32_to_bf16_rne(in[i]);
}

// ---------------- weight convert + K-pair pack: [K,N] f32 -> [K/2,N] u32 ----------------
__global__ void pb_cvt_pack_bf16(const float* __restrict__ in,
                                 unsigned* __restrict__ out, int K, int N) {
    size_t i = (size_t)blockIdx.x * 256 + threadIdx.x;
    size_t total = (size_t)(K / 2) * N;
    if (i >= total) return;
    int n  = (int)(i % N);
    int kp = (int)(i / N);
    unsigned short lo = f32_to_bf16_rne(in[(size_t)(2 * kp) * N + n]);
    unsigned short hi = f32_to_bf16_rne(in[(size_t)(2 * kp + 1) * N + n]);
    out[i] = (unsigned)lo | ((unsigned)hi << 16);
}

// ---------------- embedding + positional encoding ----------------
__global__ void pb_embed(const int* __restrict__ ids,
                         const float* __restrict__ ew,
                         const float* __restrict__ eb,
                         float* __restrict__ X) {
    size_t idx = (size_t)blockIdx.x * 256 + threadIdx.x;
    if (idx >= (size_t)ROWS * DMODEL) return;
    int d  = (int)(idx % DMODEL);
    int bs = (int)(idx / DMODEL);
    int s  = bs % SEQ;
    float ang = (float)s / __powf(10000.0f, (2.0f * (float)(d / 2)) / (float)DMODEL);
    float pos = (d & 1) ? __cosf(ang) : __sinf(ang);
    X[idx] = (float)ids[bs] * ew[d] + eb[d] + pos;
}

// ---------------- (p x p) avg pool + layernorm -> NXB bf16 [B*Sp, Dp] ----------------
__global__ void pb_pool_ln(const float* __restrict__ X,
                           const float* __restrict__ g,
                           const float* __restrict__ bta,
                           unsigned short* __restrict__ NXB,
                           int Sp, int p, int Dp) {
    int row = blockIdx.x;               // b*Sp + sp
    int b = row / Sp, sp = row % Sp;
    __shared__ float pooled[DMODEL];
    __shared__ float red[256];
    float inv_pp = 1.0f / (float)(p * p);
    for (int dp = threadIdx.x; dp < Dp; dp += 256) {
        float s = 0.0f;
        for (int i = 0; i < p; i++)
            for (int j = 0; j < p; j++)
                s += X[((size_t)(b * SEQ + sp * p + i)) * DMODEL + dp * p + j];
        pooled[dp] = s * inv_pp;
    }
    __syncthreads();
    float s = 0.0f;
    for (int dp = threadIdx.x; dp < Dp; dp += 256) s += pooled[dp];
    float mean = blockReduceSum(s, red) / (float)Dp;
    float v = 0.0f;
    for (int dp = threadIdx.x; dp < Dp; dp += 256) {
        float d0 = pooled[dp] - mean;
        v += d0 * d0;
    }
    float var = blockReduceSum(v, red) / (float)Dp;
    float inv = rsqrtf(var + 1e-6f);
    for (int dp = threadIdx.x; dp < Dp; dp += 256)
        NXB[(size_t)row * Dp + dp] =
            f32_to_bf16_rne((pooled[dp] - mean) * inv * g[dp] + bta[dp]);
}

// ---------------- pooled mask ----------------
__global__ void pb_pool_mask(const float* __restrict__ mask,
                             float* __restrict__ pm, int Sp, int p) {
    int i = blockIdx.x * 256 + threadIdx.x;
    if (i >= BATCH * Sp) return;
    int b = i / Sp, sp = i % Sp;
    float s = 0.0f;
    for (int j = 0; j < p; j++) s += mask[b * SEQ + sp * p + j];
    pm[i] = s / (float)p;
}

// ---------------- per-head attention -> bf16 (raw-reshape indexing) ----------------
__global__ void pb_attn(const float* __restrict__ Q,
                        const float* __restrict__ Km,
                        const float* __restrict__ V,
                        const float* __restrict__ pm,
                        unsigned short* __restrict__ ATTB, int Sp) {
    int idx = blockIdx.x;               // ((h*B)+b)*Sp + sq
    int sq = idx % Sp;
    int hb = idx / Sp;
    int b  = hb % BATCH;
    __shared__ float qv[64];
    __shared__ float logits[SEQ];
    __shared__ float red[256];
    __shared__ float acc4[4][64];
    const size_t base = (size_t)hb * Sp * 64;
    if (threadIdx.x < 64)
        qv[threadIdx.x] = Q[base + (size_t)sq * 64 + threadIdx.x] * pm[b * Sp + sq];
    __syncthreads();
    float rs = rsqrtf((float)DMODEL);
    for (int j = threadIdx.x; j < Sp; j += 256) {
        const float* kr = &Km[base + (size_t)j * 64];
        float dot = 0.0f;
        for (int d = 0; d < 64; d++) dot += qv[d] * kr[d];
        logits[j] = dot * rs;
    }
    __syncthreads();
    float m = -INFINITY;
    for (int j = threadIdx.x; j < Sp; j += 256) m = fmaxf(m, logits[j]);
    m = blockReduceMax(m, red);
    float s = 0.0f;
    for (int j = threadIdx.x; j < Sp; j += 256) {
        float e = __expf(logits[j] - m);
        logits[j] = e;
        s += e;
    }
    s = blockReduceSum(s, red);
    float invs = 1.0f / s;
    for (int j = threadIdx.x; j < Sp; j += 256) logits[j] *= invs;
    __syncthreads();
    // weighted sum over keys: all 256 threads (4 key-slices x 64 dims)
    {
        int q4 = threadIdx.x >> 6;
        int d  = threadIdx.x & 63;
        float acc = 0.0f;
        for (int j = q4; j < Sp; j += 4)
            acc += logits[j] * V[base + (size_t)j * 64 + d];
        acc4[q4][d] = acc;
    }
    __syncthreads();
    if (threadIdx.x < 64) {
        int d = threadIdx.x;
        float acc = acc4[0][d] + acc4[1][d] + acc4[2][d] + acc4[3][d];
        ATTB[base + (size_t)sq * 64 + d] = f32_to_bf16_rne(acc);
    }
}

// ---------------- upsample + residual + layernorm -> NXB bf16 [ROWS, DMODEL] ----------------
__global__ void pb_up_ln(const float* __restrict__ X,
                         const float* __restrict__ L,     // [B, Sp, Dp] f32
                         const float* __restrict__ g,
                         const float* __restrict__ bta,
                         unsigned short* __restrict__ NXB, int p, int Dp) {
    int row = blockIdx.x;               // b*SEQ + s
    int b = row / SEQ, s = row % SEQ;
    int sp = s / p;
    __shared__ float t[DMODEL];
    __shared__ float red[256];
    for (int d = threadIdx.x; d < DMODEL; d += 256)
        t[d] = X[(size_t)row * DMODEL + d] +
               L[((size_t)(b * (SEQ / p) + sp)) * Dp + d / p];
    __syncthreads();
    float sum = 0.0f;
    for (int d = threadIdx.x; d < DMODEL; d += 256) sum += t[d];
    float mean = blockReduceSum(sum, red) / (float)DMODEL;
    float v = 0.0f;
    for (int d = threadIdx.x; d < DMODEL; d += 256) {
        float d0 = t[d] - mean;
        v += d0 * d0;
    }
    float var = blockReduceSum(v, red) / (float)DMODEL;
    float inv = rsqrtf(var + 1e-6f);
    for (int d = threadIdx.x; d < DMODEL; d += 256)
        NXB[(size_t)row * DMODEL + d] =
            f32_to_bf16_rne((t[d] - mean) * inv * g[d] + bta[d]);
}

// ---------------- WMMA bf16 GEMM: TDM-staged, double-buffered, fused epilogue ----------------
// out = act(A_bf16[M,K] @ B[K,N] + bias) + residual ; out -> Cf (f32) or Cb (bf16)
__global__ __launch_bounds__(256)
void pb_gemm_wmma(const unsigned short* __restrict__ A,
                  const unsigned* __restrict__ Bp,     // pair-packed [K/2, N]
                  const float* __restrict__ bias,
                  const float* __restrict__ residual,
                  float* __restrict__ Cf,
                  unsigned short* __restrict__ Cb,
                  int M, int N, int K, int act) {
    __shared__ unsigned short As[2][BM][BKP];       // 2 x 9.0 KB (TDM pads rows to 36 u16)
    __shared__ unsigned       Bsp[2][BK / 2][BNP];  // 2 x 8.1 KB (TDM pads rows to 130 u32)

    const int tid  = threadIdx.x;
    const int wave = tid >> 5;              // 0..7
    const int lane = tid & 31;
    const int wrow = wave & 1;              // 2 waves along M
    const int wcol = wave >> 1;             // 4 waves along N
    const int blockM = blockIdx.y * BM;
    const int blockN = blockIdx.x * BN;
    const int m16 = lane & 15;
    const int hi  = lane >> 4;
    const int K2  = K >> 1;

    f32x8 acc[4][2];
#pragma unroll
    for (int mr = 0; mr < 4; mr++)
#pragma unroll
        for (int nc = 0; nc < 2; nc++)
#pragma unroll
            for (int v = 0; v < 8; v++) acc[mr][nc][v] = 0.0f;

    // TDM issue for K-tile at k0 into buffer `buf` (wave 0 only; EXEC ignored by TDM)
    auto issue_tile = [&](int buf, int k0) {
        // A tile: BM x BK u16, tensor extents = remaining from tile start (HW zero-fills OOB)
        tdm_load_2d(lds_addr_u32(&As[buf][0][0]),
                    A + (size_t)blockM * K + k0,
                    K - k0, M - blockM, BK, BM, K,
                    /*dsz=2B*/1, /*pad after 16 dwords*/3);
        // B tile: (BK/2) x BN dwords from pair-packed global
        tdm_load_2d(lds_addr_u32(&Bsp[buf][0][0]),
                    Bp + (size_t)(k0 >> 1) * N + blockN,
                    N - blockN, K2 - (k0 >> 1), BN, BK / 2, N,
                    /*dsz=4B*/2, /*pad after 128 dwords*/6);
    };

    if (wave == 0) issue_tile(0, 0);

    for (int t = 0, k0 = 0; k0 < K; t++, k0 += BK) {
        int cur = t & 1;
        __builtin_amdgcn_s_wait_tensorcnt(0);   // tile `t` resident in LDS (no-op for waves 1..7)
        __syncthreads();
        if (wave == 0 && k0 + BK < K) issue_tile(cur ^ 1, k0 + BK);  // overlap DMA with WMMA

        // A fragments: one ds_load_b32 per packed bf16 pair
        u32x8 afrag[4];
#pragma unroll
        for (int mr = 0; mr < 4; mr++) {
            int row = wrow * 64 + mr * 16 + m16;
            const unsigned* As32 = (const unsigned*)&As[cur][row][0];
#pragma unroll
            for (int v = 0; v < 8; v++) {
                int kk = (v < 4) ? (hi * 8 + 2 * v) : (16 + hi * 8 + 2 * (v - 4));
                afrag[mr][v] = As32[kk >> 1];
            }
        }
        // B fragments: pair-packed rows -> one ds_load_b32 per dword
        u32x8 bfrag[2];
#pragma unroll
        for (int nc = 0; nc < 2; nc++) {
            int col = wcol * 32 + nc * 16 + m16;
#pragma unroll
            for (int v = 0; v < 8; v++)
                bfrag[nc][v] = Bsp[cur][(v < 4) ? (hi * 4 + v) : (8 + hi * 4 + (v - 4))][col];
        }
#pragma unroll
        for (int mr = 0; mr < 4; mr++) {
            bf16x16 av = __builtin_bit_cast(bf16x16, afrag[mr]);
#pragma unroll
            for (int nc = 0; nc < 2; nc++) {
                bf16x16 bv = __builtin_bit_cast(bf16x16, bfrag[nc]);
                acc[mr][nc] = __builtin_amdgcn_wmma_f32_16x16x32_bf16(
                    false, av, false, bv, (short)0, acc[mr][nc], false, false);
            }
        }
        // next iteration's barrier (post tensor-wait) protects buffer reuse
    }

    // epilogue: D layout — lane n = col, VGPR v + 8*hi = row
#pragma unroll
    for (int mr = 0; mr < 4; mr++)
#pragma unroll
        for (int nc = 0; nc < 2; nc++)
#pragma unroll
            for (int v = 0; v < 8; v++) {
                int row = blockM + wrow * 64 + mr * 16 + v + 8 * hi;
                int col = blockN + wcol * 32 + nc * 16 + m16;
                if (row < M && col < N) {
                    float x = acc[mr][nc][v];
                    if (bias)     x += bias[col];
                    if (act == 1) x = (x > 0.0f) ? x : 0.3f * x;   // leaky_relu(0.3)
                    if (residual) x += residual[(size_t)row * N + col];
                    if (Cf) Cf[(size_t)row * N + col] = x;
                    else    Cb[(size_t)row * N + col] = f32_to_bf16_rne(x);
                }
            }
}

// ---------------- row softmax over VOCAB (in place on d_out) ----------------
__global__ void pb_softmax_rows(float* __restrict__ data, int cols) {
    int row = blockIdx.x;
    __shared__ float red[256];
    float* r = data + (size_t)row * cols;
    float m = -INFINITY;
    for (int c = threadIdx.x; c < cols; c += 256) m = fmaxf(m, r[c]);
    m = blockReduceMax(m, red);
    float s = 0.0f;
    for (int c = threadIdx.x; c < cols; c += 256) {
        float e = __expf(r[c] - m);
        r[c] = e;
        s += e;
    }
    s = blockReduceSum(s, red);
    float inv = 1.0f / s;
    for (int c = threadIdx.x; c < cols; c += 256) r[c] *= inv;
}

// ---------------- host orchestration ----------------
extern "C" void kernel_launch(void* const* d_in, const int* in_sizes, int n_in,
                              void* d_out, int out_size, void* d_ws, size_t ws_size,
                              hipStream_t stream) {
    const int*   ids     = (const int*)d_in[0];
    const float* mask    = (const float*)d_in[1];
    const float* embed_w = (const float*)d_in[2];
    const float* embed_b = (const float*)d_in[3];
    const float* mlm_w   = (const float*)d_in[4];
    const float* mlm_b   = (const float*)d_in[5];

    // carve workspace
    char* wp = (char*)d_ws;
    auto take = [&](size_t bytes) {
        void* p = (void*)wp;
        wp += (bytes + 255) & ~(size_t)255;
        return p;
    };
    float*          X    = (float*)take((size_t)ROWS * DMODEL * 4);
    unsigned short* NXB  = (unsigned short*)take((size_t)ROWS * DMODEL * 2);
    float*          Qb   = (float*)take((size_t)ROWS * DMODEL * 4);
    float*          Kb   = (float*)take((size_t)ROWS * DMODEL * 4);
    float*          Vb   = (float*)take((size_t)ROWS * DMODEL * 4);
    unsigned short* ATTB = (unsigned short*)take((size_t)ROWS * DMODEL * 2);
    float*          LIN  = (float*)take((size_t)ROWS * DMODEL * 4);
    unsigned short* HB   = (unsigned short*)take((size_t)ROWS * FFN * 2);
    float*          PM   = (float*)take((size_t)ROWS * 4);
    unsigned short* ABF  = (unsigned short*)take((size_t)ROWS * DMODEL * 2);
    unsigned*       BPK  = (unsigned*)take((size_t)(DMODEL / 2) * VOCAB * 4); // 49.15 MB max

    auto cvt = [&](const float* src, unsigned short* dst, size_t n) {
        pb_cvt_bf16<<<(unsigned)((n + 255) / 256), 256, 0, stream>>>(src, dst, n);
    };
    auto pack = [&](const float* src, int K, int N) {
        size_t total = (size_t)(K / 2) * N;
        pb_cvt_pack_bf16<<<(unsigned)((total + 255) / 256), 256, 0, stream>>>(src, BPK, K, N);
    };
    auto gemm = [&](const unsigned short* A, const float* bias, const float* resid,
                    float* Cf, unsigned short* Cb, int M, int N, int K, int act) {
        dim3 g((N + BN - 1) / BN, (M + BM - 1) / BM);
        pb_gemm_wmma<<<g, 256, 0, stream>>>(A, BPK, bias, resid, Cf, Cb, M, N, K, act);
    };

    // embedding + positional encoding
    pb_embed<<<(ROWS * DMODEL + 255) / 256, 256, 0, stream>>>(ids, embed_w, embed_b, X);

    const int pools[5] = {16, 8, 4, 2, 1};
    for (int bi = 0; bi < 5; bi++) {
        int p  = pools[bi];
        int Dp = DMODEL / p;
        int Sp = SEQ / p;
        int Mr = BATCH * Sp;
        int base = 6 + bi * 16;
        const float* wq_w = (const float*)d_in[base + 0];
        const float* wq_b = (const float*)d_in[base + 1];
        const float* wk_w = (const float*)d_in[base + 2];
        const float* wk_b = (const float*)d_in[base + 3];
        const float* wv_w = (const float*)d_in[base + 4];
        const float* wv_b = (const float*)d_in[base + 5];
        const float* lin_w = (const float*)d_in[base + 6];
        const float* lin_b = (const float*)d_in[base + 7];
        const float* ln1_g = (const float*)d_in[base + 8];
        const float* ln1_b = (const float*)d_in[base + 9];
        const float* ln2_g = (const float*)d_in[base + 10];
        const float* ln2_b = (const float*)d_in[base + 11];
        const float* f1_w  = (const float*)d_in[base + 12];
        const float* f1_b  = (const float*)d_in[base + 13];
        const float* f2_w  = (const float*)d_in[base + 14];
        const float* f2_b  = (const float*)d_in[base + 15];

        pb_pool_mask<<<(Mr + 255) / 256, 256, 0, stream>>>(mask, PM, Sp, p);
        pb_pool_ln<<<Mr, 256, 0, stream>>>(X, ln1_g, ln1_b, NXB, Sp, p, Dp);

        // QKV projections: [Mr, Dp] x [Dp, 768] -> f32
        pack(wq_w, Dp, DMODEL);
        gemm(NXB, wq_b, nullptr, Qb, nullptr, Mr, DMODEL, Dp, 0);
        pack(wk_w, Dp, DMODEL);
        gemm(NXB, wk_b, nullptr, Kb, nullptr, Mr, DMODEL, Dp, 0);
        pack(wv_w, Dp, DMODEL);
        gemm(NXB, wv_b, nullptr, Vb, nullptr, Mr, DMODEL, Dp, 0);

        // attention (small, VALU path) -> bf16 A operand for lin
        pb_attn<<<NHEADS * BATCH * Sp, 256, 0, stream>>>(Qb, Kb, Vb, PM, ATTB, Sp);

        // output projection: [Mr, 768] x [768, Dp] -> f32 LIN
        pack(lin_w, DMODEL, Dp);
        gemm(ATTB, lin_b, nullptr, LIN, nullptr, Mr, Dp, DMODEL, 0);

        // upsample + residual + LN2 -> bf16 NXB
        pb_up_ln<<<ROWS, 256, 0, stream>>>(X, LIN, ln2_g, ln2_b, NXB, p, Dp);

        // FFN1: leaky_relu(n2 @ f1 + b) -> bf16 HB directly
        pack(f1_w, DMODEL, FFN);
        gemm(NXB, f1_b, nullptr, nullptr, HB, ROWS, FFN, DMODEL, 1);

        // FFN2 + residual from block input (in-place on X, f32)
        pack(f2_w, FFN, DMODEL);
        gemm(HB, f2_b, X, X, nullptr, ROWS, DMODEL, FFN, 0);
    }

    // MLM head -> d_out, then softmax in place
    cvt(X, ABF, (size_t)ROWS * DMODEL);
    pack(mlm_w, DMODEL, VOCAB);
    gemm(ABF, mlm_b, nullptr, (float*)d_out, nullptr, ROWS, VOCAB, DMODEL, 0);
    pb_softmax_rows<<<ROWS, 256, 0, stream>>>((float*)d_out, VOCAB);
}
